// GraphNodeTimeSeriesEncoder_83794811945451
// MI455X (gfx1250) — matrix operand order repaired and checked by reading, not compiled, not run on hardware
//
#include <hip/hip_runtime.h>
#include <hip/hip_bf16.h>

// ---------------------------------------------------------------------------
// GraphNodeTimeSeriesEncoder for MI455X (gfx1250, wave32, WMMA bf16)
// out[n,t,h] = relu( LN(feat[n,t,:]) . W[h,:] + b[h] )
// GEMM [131072 x 545] x [545 x 1024] via v_wmma_f32_16x16x32_bf16, K padded 576
// B tiles double-buffered in LDS via global_load_async_to_lds_b128 (ASYNCcnt)
// ---------------------------------------------------------------------------

#define LATENT   256
#define TPATCH   16
#define TOUT     256
#define HIDDEN   1024
#define IN_DIM   545
#define KPAD     576          // 18 k-tiles of 32
#define KTILES   18
#define CTILES   8            // 1024 cols / 128
#define BSIZE    8
#define NNODES   512
#define ROWS_PB  128          // rows (n,t pairs) per workgroup
#define THREADS  256          // 8 wave32

#define A_BYTES        (ROWS_PB * KPAD * 2)   // 147456
#define B_STAGE_BYTES  (128 * 64)             // 8KB: 128 cols x 32 K bf16

typedef __attribute__((ext_vector_type(16))) __bf16 v16bf;
typedef __attribute__((ext_vector_type(8)))  __bf16 v8bf;
typedef __attribute__((ext_vector_type(8)))  float  v8f;

union V16U { v16bf v; v8bf h[2]; };

__device__ __forceinline__ unsigned short f2bf_u16(float f) {
  // round-to-nearest-even f32 -> bf16 (inputs are finite normals)
  unsigned int u = __float_as_uint(f);
  u += 0x7FFFu + ((u >> 16) & 1u);
  return (unsigned short)(u >> 16);
}

// LDS aperture flat address: low 32 bits == LDS byte offset (ISA 10.2)
__device__ __forceinline__ unsigned lds_off(const void* p) {
  return (unsigned)(unsigned long long)p;
}

// Async DMA: 16B per lane, global -> LDS, tracked by ASYNCcnt (no VGPR data)
__device__ __forceinline__ void async_copy_b128(unsigned lds_byte_off,
                                                unsigned long long gaddr) {
  asm volatile("global_load_async_to_lds_b128 %0, %1, off"
               :: "v"(lds_byte_off), "v"(gaddr) : "memory");
}

__device__ __forceinline__ void wait_async_le2() {
#if __has_builtin(__builtin_amdgcn_s_wait_asynccnt)
  __builtin_amdgcn_s_wait_asynccnt(2);
#else
  asm volatile("s_wait_asynccnt 0x2" ::: "memory");
#endif
}
__device__ __forceinline__ void wait_async_le0() {
#if __has_builtin(__builtin_amdgcn_s_wait_asynccnt)
  __builtin_amdgcn_s_wait_asynccnt(0);
#else
  asm volatile("s_wait_asynccnt 0x0" ::: "memory");
#endif
}

// --- prep: W [1024 x 545] f32 -> Wb [1024 x 576] bf16 (zero padded K) -------
__global__ void wprep_kernel(const float* __restrict__ W, unsigned short* __restrict__ Wb) {
  const int h = blockIdx.x;              // 0..1023
  for (int k = threadIdx.x; k < KPAD; k += blockDim.x) {
    float v = (k < IN_DIM) ? W[(size_t)h * IN_DIM + k] : 0.0f;
    Wb[(size_t)h * KPAD + k] = f2bf_u16(v);
  }
}

// --- fused feature-build + LayerNorm + async-staged bf16 WMMA GEMM ----------
__global__ __launch_bounds__(THREADS, 1)
void encoder_wmma_kernel(const float* __restrict__ gx,   // [8,256,16]
                         const float* __restrict__ gy,   // [8,256,16]
                         const float* __restrict__ ne,   // [512,256]
                         const float* __restrict__ lat,  // [8,256]
                         const int*   __restrict__ ptr,  // [9]
                         const float* __restrict__ gamma,// [545]
                         const float* __restrict__ beta, // [545]
                         const unsigned short* __restrict__ Wb, // [1024,576] bf16
                         const float* __restrict__ bias, // [1024]
                         float* __restrict__ out)        // [131072,1024]
{
  extern __shared__ __align__(64) unsigned char smem[];
  unsigned short* Alds  = (unsigned short*)smem;                        // 128x576 bf16
  unsigned short* Bbuf0 = (unsigned short*)(smem + A_BYTES);            // 8KB stage
  unsigned short* Bbuf1 = (unsigned short*)(smem + A_BYTES + B_STAGE_BYTES);
  float* fs    = (float*)(smem + A_BYTES + 2 * B_STAGE_BYTES);
  float* sfeat = fs;            // 512  shared features (node_emb | latent)
  float* sg    = fs + 512;      // 576  gamma (0-padded)
  float* sb    = fs + 1088;     // 576  beta  (0-padded)
  float* smu   = fs + 1664;     // 128  per-row mean
  float* srs   = fs + 1792;     // 128  per-row rsqrt(var+eps)
  float* sred  = fs + 1920;     // 512  reduction scratch (sum | sumsq)

  const int tid = threadIdx.x;
  const int tileRow = blockIdx.x;            // 0..1023
  const int n  = tileRow >> 1;               // node id
  const int t0 = (tileRow & 1) * ROWS_PB;    // first timestep of tile

  // ---- per-thread async source/dest mapping: 2 x b128 per thread/stage ----
  // stage = 128 cols x 64B; idx = tid + i*256 -> col = idx/4, quarter = idx%4
  unsigned gthr[2], lthr[2];
  #pragma unroll
  for (int i = 0; i < 2; ++i) {
    const int idx = tid + i * THREADS;
    const int col = idx >> 2, q = idx & 3;
    gthr[i] = (unsigned)col * (KPAD * 2) + (unsigned)q * 16;  // bytes into Wb row
    lthr[i] = (unsigned)col * 64 + (unsigned)q * 16;          // bytes into stage
  }
  const unsigned long long wbB = (unsigned long long)Wb;
  const unsigned bbase[2] = { lds_off(Bbuf0), lds_off(Bbuf1) };

  // kick off stage (ct=0, kt=0) so the copy overlaps feature build + LN
  #pragma unroll
  for (int i = 0; i < 2; ++i)
    async_copy_b128(bbase[0] + lthr[i], wbB + gthr[i]);

  // CSR searchsorted(ptr, n, 'right') - 1
  int bidx = 0;
  #pragma unroll
  for (int i = 1; i < BSIZE; ++i) if (ptr[i] <= n) bidx = i;
  const size_t gmb = (size_t)bidx * TOUT * TPATCH;

  // ---- stage shared features + gamma/beta -------------------------------
  sfeat[tid]       = ne[(size_t)n * LATENT + tid];
  sfeat[256 + tid] = lat[(size_t)bidx * LATENT + tid];
  for (int k = tid; k < KPAD; k += THREADS) {
    sg[k] = (k < IN_DIM) ? gamma[k] : 0.0f;
    sb[k] = (k < IN_DIM) ? beta[k]  : 0.0f;
  }
  __syncthreads();

  // ---- shared-part sum / sumsq (deterministic tree reduction) -----------
  {
    float a0 = sfeat[tid], a1 = sfeat[256 + tid];
    sred[tid]       = a0 + a1;
    sred[256 + tid] = a0 * a0 + a1 * a1;
  }
  __syncthreads();
  for (int off = 128; off > 0; off >>= 1) {
    if (tid < off) {
      sred[tid]       += sred[tid + off];
      sred[256 + tid] += sred[256 + tid + off];
    }
    __syncthreads();
  }

  // ---- per-row LN statistics (shared part reused across 128 rows) -------
  if (tid < ROWS_PB) {
    const int t = t0 + tid;
    float s = 0.0f, q = 0.0f;
    #pragma unroll
    for (int j = 0; j < TPATCH; ++j) {
      float vx = gx[gmb + (size_t)t * TPATCH + j];
      float vy = gy[gmb + (size_t)t * TPATCH + j];
      s += vx + vy;
      q += vx * vx + vy * vy;
    }
    const float tf = (float)t * (1.0f / TOUT);
    s += tf; q += tf * tf;
    const float S = sred[0] + s;
    const float Q = sred[256] + q;
    const float mu = S * (1.0f / IN_DIM);
    const float var = Q * (1.0f / IN_DIM) - mu * mu;
    smu[tid] = mu;
    srs[tid] = rsqrtf(var + 1e-5f);
  }
  __syncthreads();

  // ---- build normalized bf16 A tile [128 x 576] in LDS ------------------
  for (int idx = tid; idx < ROWS_PB * KPAD; idx += THREADS) {
    const int row = idx / KPAD;
    const int k   = idx - row * KPAD;
    const int t   = t0 + row;
    float f;
    if (k < 512)            f = sfeat[k];
    else if (k < 528)       f = gx[gmb + (size_t)t * TPATCH + (k - 512)];
    else if (k < 544)       f = gy[gmb + (size_t)t * TPATCH + (k - 528)];
    else if (k == 544)      f = (float)t * (1.0f / TOUT);
    else { Alds[idx] = 0; continue; }                       // K padding
    const float x = (f - smu[row]) * srs[row] * sg[k] + sb[k];
    Alds[idx] = f2bf_u16(x);
  }

  // ---- WMMA GEMM: 8 waves tile 128 rows x 128 cols, loop 8 col-tiles ----
  const int wid  = tid >> 5;          // wave id 0..7 (wave32)
  const int lane = tid & 31;
  const int wr   = wid >> 1;          // 0..3 : 32-row strip
  const int wc   = wid & 1;           // 0..1 : 64-col strip
  const int hh   = lane >> 4;         // lane half
  const int l16  = lane & 15;
  const size_t rowBase = (size_t)n * TOUT + t0;
  unsigned short* Bloc[2] = { Bbuf0, Bbuf1 };

  for (int ct = 0; ct < CTILES; ++ct) {
    v8f acc[2][4];
    #pragma unroll
    for (int ms = 0; ms < 2; ++ms)
      #pragma unroll
      for (int cn = 0; cn < 4; ++cn) {
        v8f z = {0.f,0.f,0.f,0.f,0.f,0.f,0.f,0.f};
        acc[ms][cn] = z;
      }

    for (int kt = 0; kt < KTILES; ++kt) {
      // all waves done reading buf[(kt+1)&1] (last read in the previous stage)
      __syncthreads();
      const bool last = (ct == CTILES - 1) && (kt == KTILES - 1);
      if (!last) {
        const int nct = (kt == KTILES - 1) ? ct + 1 : ct;
        const int nkt = (kt == KTILES - 1) ? 0 : kt + 1;
        const unsigned goff = (unsigned)nct * (128u * KPAD * 2u) + (unsigned)nkt * 64u;
        #pragma unroll
        for (int i = 0; i < 2; ++i)
          async_copy_b128(bbase[(kt + 1) & 1] + lthr[i], wbB + goff + gthr[i]);
        wait_async_le2();   // stage (ct,kt) complete; (next) still in flight
      } else {
        wait_async_le0();
      }
      __syncthreads();      // stage (ct,kt) visible to all waves

      // A fragments: 16-bit A 16x32 layout -> lane l16 = M,
      // halfs[0..7] = K(kt*32 + hh*8 ..), halfs[8..15] = K(kt*32+16+hh*8 ..)
      v16bf afrag[2];
      #pragma unroll
      for (int ms = 0; ms < 2; ++ms) {
        const unsigned short* ap =
            Alds + (size_t)(wr * 32 + ms * 16 + l16) * KPAD + kt * 32;
        V16U u;
        u.h[0] = *(const v8bf*)(ap + hh * 8);
        u.h[1] = *(const v8bf*)(ap + 16 + hh * 8);
        afrag[ms] = u.v;
      }
      // B fragments from LDS stage: [col][32 K] rows of 64B; lane l16 = N,
      // halfs[0..7] = K(kt*32 + hh*8 ..), halfs[8..15] = K(kt*32+16+hh*8 ..)
      #pragma unroll
      for (int cn = 0; cn < 4; ++cn) {
        const unsigned short* bp =
            Bloc[kt & 1] + (size_t)(wc * 64 + cn * 16 + l16) * 32;
        V16U ub;
        ub.h[0] = *(const v8bf*)(bp + hh * 8);
        ub.h[1] = *(const v8bf*)(bp + 16 + hh * 8);
        const v16bf bfrag = ub.v;
        acc[0][cn] = __builtin_amdgcn_wmma_f32_16x16x32_bf16(
            false, afrag[0], false, bfrag, (short)0, acc[0][cn], false, false);
        acc[1][cn] = __builtin_amdgcn_wmma_f32_16x16x32_bf16(
            false, afrag[1], false, bfrag, (short)0, acc[1][cn], false, false);
      }
    }

    // ---- bias + ReLU + store (D layout: M = r + 8*hh, N = l16) ----------
    #pragma unroll
    for (int cn = 0; cn < 4; ++cn) {
      const int col = ct * 128 + wc * 64 + cn * 16 + l16;
      const float bc = bias[col];
      #pragma unroll
      for (int ms = 0; ms < 2; ++ms) {
        #pragma unroll
        for (int r = 0; r < 8; ++r) {
          const size_t grow = rowBase + (size_t)(wr * 32 + ms * 16 + r + 8 * hh);
          const float v = acc[ms][cn][r] + bc;
          out[grow * HIDDEN + col] = v > 0.0f ? v : 0.0f;
        }
      }
    }
  }
}

// ---------------------------------------------------------------------------
extern "C" void kernel_launch(void* const* d_in, const int* in_sizes, int n_in,
                              void* d_out, int out_size, void* d_ws, size_t ws_size,
                              hipStream_t stream) {
  const float* gx    = (const float*)d_in[0];
  const float* gy    = (const float*)d_in[1];
  const float* ne    = (const float*)d_in[2];
  const float* lat   = (const float*)d_in[3];
  const int*   ptr   = (const int*)  d_in[4];
  const float* gamma = (const float*)d_in[5];
  const float* beta  = (const float*)d_in[6];
  const float* W     = (const float*)d_in[7];
  const float* bias  = (const float*)d_in[8];
  float* out = (float*)d_out;
  unsigned short* Wb = (unsigned short*)d_ws;     // 1024*576*2 = 1.18 MB

  // 1) W -> bf16 (K-padded), written fully every call (d_ws is poisoned)
  wprep_kernel<<<dim3(HIDDEN), dim3(256), 0, stream>>>(W, Wb);

  // 2) fused LN + async-staged WMMA GEMM
  constexpr size_t LDS_BYTES =
      (size_t)A_BYTES + 2 * B_STAGE_BYTES +
      sizeof(float) * (512 + 576 + 576 + 128 + 128 + 512);
  (void)hipFuncSetAttribute((const void*)encoder_wmma_kernel,
                            hipFuncAttributeMaxDynamicSharedMemorySize,
                            (int)LDS_BYTES);
  encoder_wmma_kernel<<<dim3((NNODES * TOUT) / ROWS_PB), dim3(THREADS), LDS_BYTES, stream>>>(
      gx, gy, ne, lat, ptr, gamma, beta, Wb, bias, out);
}
